// MultiheadSelfAttention_73615739453744
// MI455X (gfx1250) — matrix operand order repaired and checked by reading, not compiled, not run on hardware
//
#include <hip/hip_runtime.h>

// Multi-head self-attention for MI455X (gfx1250, wave32, WMMA).
// Pipeline: f32->f16 convert, W transpose, 3x WMMA GEMM (Q/K/V), fused
// flash-attention with v_wmma_f32_16x16x32_f16 and f32 accumulation.
// Round 2: 32-key blocks (no K-dim zero padding in PV), global prefetch.

typedef _Float16 half_t;
typedef __attribute__((ext_vector_type(8)))  _Float16 v8h;
typedef __attribute__((ext_vector_type(16))) _Float16 v16h;
typedef __attribute__((ext_vector_type(8)))  float    v8f;

#define BATCH  2
#define S_LEN  2048
#define DMODEL 1024
#define NH     16
#define HD     64

// ---- A-operand (16x32 f16): lane holds row M=lane%16.
// halves: K = hh*16+e (contiguous, matches B operand)
static __device__ __forceinline__ v16h load_a16(const half_t* __restrict__ row,
                                                int k0, int hh) {
  v8h lo = *(const v8h*)(row + k0 + hh * 16);
  v8h hi = *(const v8h*)(row + k0 + hh * 16 + 8);
  return __builtin_shufflevector(lo, hi, 0, 1, 2, 3, 4, 5, 6, 7,
                                 8, 9, 10, 11, 12, 13, 14, 15);
}

// ---- B-operand (32x16 f16): lane holds column N=lane%16,
// contiguous K block [hh*16, hh*16+16)  -> single 32B load.
static __device__ __forceinline__ v16h load_b16(const half_t* __restrict__ col,
                                                int k0, int hh) {
  return *(const v16h*)(col + k0 + hh * 16);
}

// ---------------- conversion kernels ----------------
__global__ void k_f32_to_f16(const float* __restrict__ in,
                             half_t* __restrict__ out, int n) {
  int i = blockIdx.x * blockDim.x + threadIdx.x;
  if (i < n) out[i] = (half_t)in[i];
}

// out[n][k] = (f16) in[k][n]   (DMODEL x DMODEL)
__global__ void k_transpose_f16(const float* __restrict__ in,
                                half_t* __restrict__ out) {
  int i = blockIdx.x * blockDim.x + threadIdx.x;
  int nn = i >> 10;
  int k = i & (DMODEL - 1);
  out[i] = (half_t)in[k * DMODEL + nn];
}

// ---------------- QKV projection GEMM ----------------
// C[4096 x 1024] = Xh[4096 x 1024] * WT^T + bias, tile 16x16 per wave.
// Output: transpose_v==0 -> [B,H,S,HD] f16 ; ==1 -> [B,H,HD,S] f16 (for V).
__global__ __launch_bounds__(128) void k_qkv_gemm(
    const half_t* __restrict__ xh, const half_t* __restrict__ wt,
    const float* __restrict__ bias, half_t* __restrict__ out, int transpose_v) {
  int lane = threadIdx.x & 31;
  int wv = threadIdx.x >> 5;
  int l16 = lane & 15, hh = lane >> 4;
  int tileM = blockIdx.x;              // 0..255
  int tileN = blockIdx.y * 4 + wv;     // 0..63

  const half_t* arow = xh + (size_t)(tileM * 16 + l16) * DMODEL;
  const half_t* brow = wt + (size_t)(tileN * 16 + l16) * DMODEL;

  v8f c = {};
#pragma unroll 4
  for (int k0 = 0; k0 < DMODEL; k0 += 32) {
    v16h a = load_a16(arow, k0, hh);
    v16h b = load_b16(brow, k0, hh);
    c = __builtin_amdgcn_wmma_f32_16x16x32_f16(false, a, false, b, (short)0, c,
                                               false, false);
  }

  int n = tileN * 16 + l16;
  float bs = bias[n];
  int h = n / HD, d = n % HD;
#pragma unroll
  for (int r = 0; r < 8; ++r) {
    int t = tileM * 16 + r + 8 * hh;   // token id
    int bb = t / S_LEN, s = t % S_LEN;
    half_t hv = (half_t)(c[r] + bs);
    if (!transpose_v)
      out[(((size_t)(bb * NH + h) * S_LEN) + s) * HD + d] = hv;
    else
      out[(((size_t)(bb * NH + h) * HD) + d) * S_LEN + s] = hv;
  }
}

// ---------------- fused flash attention ----------------
// One wave: 16 queries x full HD=64 output, streaming 32-key blocks.
__global__ __launch_bounds__(128) void k_attn(const half_t* __restrict__ Qh,
                                              const half_t* __restrict__ Kh,
                                              const half_t* __restrict__ Vt,
                                              float* __restrict__ out) {
  __shared__ __align__(32) half_t lds[4 * 512];  // 16x32 f16 P-tile per wave
  int lane = threadIdx.x & 31;
  int wv = threadIdx.x >> 5;
  int l16 = lane & 15, hh = lane >> 4;
  int bh = blockIdx.x;                 // b*NH + h
  int bb = bh >> 4, h = bh & 15;
  int qt = blockIdx.y * 4 + wv;        // query tile 0..127
  half_t* myLds = lds + wv * 512;

  const half_t* qrow = Qh + ((size_t)bh * S_LEN + qt * 16 + l16) * HD;
  v16h qa0 = load_a16(qrow, 0, hh);
  v16h qa1 = load_a16(qrow, 32, hh);

  const half_t* kbh = Kh + (size_t)bh * S_LEN * HD;
  const half_t* vbh = Vt + (size_t)bh * HD * S_LEN;

  float m[8], l[8], p0[8], p1[8];
  v8f o0 = {}, o1 = {}, o2 = {}, o3 = {};
#pragma unroll
  for (int r = 0; r < 8; ++r) {
    m[r] = -__builtin_inff();
    l[r] = 0.0f;
  }

  for (int kt = 0; kt < S_LEN / 32; ++kt) {
    // ---- scores: two 16x16 tiles covering keys [kt*32, kt*32+32)
    const half_t* krow0 = kbh + (size_t)(kt * 32 + l16) * HD;
    const half_t* krow1 = krow0 + 16 * HD;
    // prefetch next key block's K rows (global_prefetch_b8)
    __builtin_prefetch(krow0 + 32 * HD, 0, 1);
    __builtin_prefetch(krow1 + 32 * HD, 0, 1);
    v16h kb00 = load_b16(krow0, 0, hh);
    v16h kb01 = load_b16(krow0, 32, hh);
    v16h kb10 = load_b16(krow1, 0, hh);
    v16h kb11 = load_b16(krow1, 32, hh);
    v8f s0 = {}, s1 = {};
    s0 = __builtin_amdgcn_wmma_f32_16x16x32_f16(false, qa0, false, kb00,
                                                (short)0, s0, false, false);
    s0 = __builtin_amdgcn_wmma_f32_16x16x32_f16(false, qa1, false, kb01,
                                                (short)0, s0, false, false);
    s1 = __builtin_amdgcn_wmma_f32_16x16x32_f16(false, qa0, false, kb10,
                                                (short)0, s1, false, false);
    s1 = __builtin_amdgcn_wmma_f32_16x16x32_f16(false, qa1, false, kb11,
                                                (short)0, s1, false, false);

    // ---- online softmax over 32 columns; row r+8*hh across 16-lane group
#pragma unroll
    for (int r = 0; r < 8; ++r) {
      float sv0 = s0[r] * 0.125f;
      float sv1 = s1[r] * 0.125f;
      float mt = fmaxf(sv0, sv1);
      mt = fmaxf(mt, __shfl_xor(mt, 1, 32));
      mt = fmaxf(mt, __shfl_xor(mt, 2, 32));
      mt = fmaxf(mt, __shfl_xor(mt, 4, 32));
      mt = fmaxf(mt, __shfl_xor(mt, 8, 32));
      float mn = fmaxf(m[r], mt);
      float alpha = __expf(m[r] - mn);
      m[r] = mn;
      float pv0 = __expf(sv0 - mn);
      float pv1 = __expf(sv1 - mn);
      float rs = pv0 + pv1;
      rs += __shfl_xor(rs, 1, 32);
      rs += __shfl_xor(rs, 2, 32);
      rs += __shfl_xor(rs, 4, 32);
      rs += __shfl_xor(rs, 8, 32);
      l[r] = l[r] * alpha + rs;
      p0[r] = pv0;
      p1[r] = pv1;
      o0[r] *= alpha;
      o1[r] *= alpha;
      o2[r] *= alpha;
      o3[r] *= alpha;
    }

    // ---- C-layout -> A-operand re-layout through LDS: P is 16x32 f16
#pragma unroll
    for (int r = 0; r < 8; ++r) {
      int row = (r + 8 * hh) * 32;
      myLds[row + l16] = (half_t)p0[r];
      myLds[row + 16 + l16] = (half_t)p1[r];
    }
    __syncthreads();
    v16h pa = load_a16(myLds + l16 * 32, 0, hh);  // full K=32, no padding
    __syncthreads();

    // ---- O += P * V ; all 32 K valid, single WMMA per 16-dim chunk
    int tk0 = kt * 32;
#pragma unroll
    for (int ch = 0; ch < 4; ++ch) {
      const half_t* vrow = vbh + (size_t)(ch * 16 + l16) * S_LEN + tk0;
      __builtin_prefetch(vrow + 32, 0, 1);  // next key block's V
      v16h vb = *(const v16h*)(vrow + hh * 16);
      v8f* oc = (ch == 0) ? &o0 : (ch == 1) ? &o1 : (ch == 2) ? &o2 : &o3;
      *oc = __builtin_amdgcn_wmma_f32_16x16x32_f16(false, pa, false, vb,
                                                   (short)0, *oc, false, false);
    }
  }

  // ---- normalize and write fp32 output [B,S,H*HD]
#pragma unroll
  for (int r = 0; r < 8; ++r) {
    int q = qt * 16 + r + 8 * hh;
    float inv = 1.0f / l[r];
    size_t base = ((size_t)bb * S_LEN + q) * DMODEL + h * HD + l16;
    out[base + 0] = o0[r] * inv;
    out[base + 16] = o1[r] * inv;
    out[base + 32] = o2[r] * inv;
    out[base + 48] = o3[r] * inv;
  }
}

extern "C" void kernel_launch(void* const* d_in, const int* in_sizes, int n_in,
                              void* d_out, int out_size, void* d_ws,
                              size_t ws_size, hipStream_t stream) {
  const float* x = (const float*)d_in[0];
  const float* Wq = (const float*)d_in[1];
  const float* bq = (const float*)d_in[2];
  const float* Wk = (const float*)d_in[3];
  const float* bk = (const float*)d_in[4];
  const float* Wv = (const float*)d_in[5];
  const float* bv = (const float*)d_in[6];
  float* out = (float*)d_out;

  char* ws = (char*)d_ws;
  half_t* xh = (half_t*)ws;                          // 8 MB
  half_t* wqt = (half_t*)(ws + (8u << 20));          // 2 MB
  half_t* wkt = wqt + (1u << 20);                    // 2 MB
  half_t* wvt = wkt + (1u << 20);                    // 2 MB
  half_t* Qh = (half_t*)(ws + (14u << 20));          // 8 MB [B,H,S,HD]
  half_t* Kh = Qh + (4u << 20);                      // 8 MB [B,H,S,HD]
  half_t* Vt = Kh + (4u << 20);                      // 8 MB [B,H,HD,S]

  int ntok = BATCH * S_LEN * DMODEL;  // 4,194,304
  k_f32_to_f16<<<ntok / 256, 256, 0, stream>>>(x, xh, ntok);
  int nw = DMODEL * DMODEL;           // 1,048,576
  k_transpose_f16<<<nw / 256, 256, 0, stream>>>(Wq, wqt);
  k_transpose_f16<<<nw / 256, 256, 0, stream>>>(Wk, wkt);
  k_transpose_f16<<<nw / 256, 256, 0, stream>>>(Wv, wvt);

  dim3 gg(256, 16), bb(128);
  k_qkv_gemm<<<gg, bb, 0, stream>>>(xh, wqt, bq, Qh, 0);
  k_qkv_gemm<<<gg, bb, 0, stream>>>(xh, wkt, bk, Kh, 0);
  k_qkv_gemm<<<gg, bb, 0, stream>>>(xh, wvt, bv, Vt, 1);

  dim3 ga(BATCH * NH, S_LEN / 64);    // 32 x 32 blocks, 4 waves each
  k_attn<<<ga, bb, 0, stream>>>(Qh, Kh, Vt, out);
}
